// MultiHeadAttention_335007449108
// MI455X (gfx1250) — matrix operand order repaired and checked by reading, not compiled, not run on hardware
//
#include <hip/hip_runtime.h>
#include <hip/hip_bf16.h>
#include <math.h>

// MI455X / gfx1250 multi-head attention, bf16 WMMA pipeline.
// Stages: (1) fp32->bf16 + weight transposes, (2) fused QKV GEMM (Q pre-scaled
// by 1/sqrt(D), V stored transposed), (3) flash-attention with online softmax,
// K/V tiles staged into LDS via async global->LDS copies (ASYNCcnt),
// (4) output projection GEMM + bias (f32 out).
// Workspace need: ~68 MB.

typedef __bf16 bf16;
typedef __attribute__((ext_vector_type(16))) __bf16 v16bf;
typedef __attribute__((ext_vector_type(8)))  __bf16 v8bf;
typedef __attribute__((ext_vector_type(8)))  float  v8f;
typedef int v4i_ __attribute__((vector_size(16)));  // matches builtin prototype

#define E_ 768
#define H_ 12
#define D_ 64
#define B_ 4
#define S_ 2048
#define BS_ (B_ * S_)   // 8192

// -------------------------------------------------------- async copy to LDS
#if __has_builtin(__builtin_amdgcn_global_load_async_to_lds_b128)
#define ASYNC_LDS 1
#else
#define ASYNC_LDS 0
#endif

#if ASYNC_LDS
#define GAS __attribute__((address_space(1)))
#define LAS __attribute__((address_space(3)))
__device__ __forceinline__ void async_copy16(const bf16* g, bf16* l) {
  // 16 bytes global -> LDS, tracked by ASYNCcnt
  __builtin_amdgcn_global_load_async_to_lds_b128(
      (GAS v4i_*)g, (LAS v4i_*)l, /*offset=*/0, /*cpol=*/0);
}
#endif

__device__ __forceinline__ void wait_async() {
#if __has_builtin(__builtin_amdgcn_s_wait_asynccnt)
  __builtin_amdgcn_s_wait_asynccnt(0);
#elif ASYNC_LDS
  asm volatile("s_wait_asynccnt 0x0" ::: "memory");
#endif
}

// ---------------------------------------------------------------- WMMA core
__device__ __forceinline__ v8f wmma_bf16(v16bf a, v16bf b, v8f c) {
  // D = A(16x32) * B(32x16) + C, f32 accumulate
  return __builtin_amdgcn_wmma_f32_16x16x32_bf16(
      /*neg_a=*/false, a, /*neg_b=*/false, b,
      /*c_mod=*/(short)0, c, /*reuse_a=*/false, /*reuse_b=*/false);
}

// A fragment (16Mx32K, bf16) from a row-major [M x ld] matrix.
// ISA layout: lane m = lane&15, elements j -> k = k0 + (j<8 ? j : j+8) + 8*(lane>=16)
__device__ __forceinline__ v16bf load_a_frag(const bf16* base, int ld, int k0) {
  const int lane = threadIdx.x & 31;
  const int m = lane & 15, hi = lane >> 4;
  const bf16* p = base + (size_t)m * ld + k0 + hi * 8;
  v8bf x0 = *(const v8bf*)(p);        // k = k0+8*hi + 0..7
  v8bf x1 = *(const v8bf*)(p + 16);   // k = k0+8*hi + 16..23
  v16bf a;
#pragma unroll
  for (int i = 0; i < 8; ++i) { a[i] = x0[i]; a[i + 8] = x1[i]; }
  return a;
}

// B fragment (32Kx16N, bf16) from a row-major [N x ld] matrix holding B^T
// (row = output column n, elements = contraction).
// ISA layout: lane n = lane&15, elements j -> k = k0 + (j<8 ? j : j+8) + 8*(lane>=16)
__device__ __forceinline__ v16bf load_b_frag(const bf16* base, int ld, int k0) {
  const int lane = threadIdx.x & 31;
  const int n = lane & 15, hi = lane >> 4;
  const bf16* p = base + (size_t)n * ld + k0 + hi * 8;
  v8bf x0 = *(const v8bf*)(p);
  v8bf x1 = *(const v8bf*)(p + 16);
  v16bf b;
#pragma unroll
  for (int i = 0; i < 8; ++i) { b[i] = x0[i]; b[i + 8] = x1[i]; }
  return b;
}

// ---------------------------------------------------------------- prep
__global__ void k_cvt_x(const float* __restrict__ x, bf16* __restrict__ xb, int n) {
  for (int i = blockIdx.x * blockDim.x + threadIdx.x; i < n; i += gridDim.x * blockDim.x)
    xb[i] = (bf16)x[i];
}

// W [H,E,D] (f32) -> WT [H,D,E] (bf16)
__global__ void k_tr_whead(const float* __restrict__ w, bf16* __restrict__ wt) {
  const int total = H_ * D_ * E_;
  for (int i = blockIdx.x * blockDim.x + threadIdx.x; i < total; i += gridDim.x * blockDim.x) {
    int e = i % E_;
    int t = i / E_;
    int d = t % D_;
    int h = t / D_;
    wt[i] = (bf16)w[((size_t)h * E_ + e) * D_ + d];
  }
}

// Wo [E,E] (f32) -> WoT [n,e] = Wo[e,n] (bf16)
__global__ void k_tr_wo(const float* __restrict__ w, bf16* __restrict__ wt) {
  const int total = E_ * E_;
  for (int i = blockIdx.x * blockDim.x + threadIdx.x; i < total; i += gridDim.x * blockDim.x) {
    int e = i % E_;
    int n = i / E_;
    wt[i] = (bf16)w[(size_t)e * E_ + n];
  }
}

// ---------------------------------------------------------------- QKV GEMM
// grid = (H, BS/128), block = 256 (8 waves, 16 M-rows each).
// Q -> q[B,H,S,D] pre-scaled by 1/sqrt(D); K -> k[B,H,S,D]; V -> vt[B,H,D,S].
__global__ __launch_bounds__(256) void k_qkv(
    const bf16* __restrict__ xb,
    const bf16* __restrict__ wqt, const bf16* __restrict__ wkt, const bf16* __restrict__ wvt,
    const float* __restrict__ bq, const float* __restrict__ bk, const float* __restrict__ bv,
    bf16* __restrict__ q, bf16* __restrict__ k, bf16* __restrict__ vt) {
  const int h = blockIdx.x;
  const int wave = threadIdx.x >> 5;
  const int lane = threadIdx.x & 31;
  const int ln = lane & 15, hi = lane >> 4;
  const int mbase = blockIdx.y * 128 + wave * 16;  // row in [0, BS)

  const bf16* wq_h = wqt + (size_t)h * D_ * E_;
  const bf16* wk_h = wkt + (size_t)h * D_ * E_;
  const bf16* wv_h = wvt + (size_t)h * D_ * E_;
  const bf16* xrow = xb + (size_t)mbase * E_;

  v8f accq[4] = {}, acck[4] = {}, accv[4] = {};
  for (int k0 = 0; k0 < E_; k0 += 32) {
    v16bf a = load_a_frag(xrow, E_, k0);
#pragma unroll
    for (int nt = 0; nt < 4; ++nt) {
      v16bf fq = load_b_frag(wq_h + (size_t)nt * 16 * E_, E_, k0);
      accq[nt] = wmma_bf16(a, fq, accq[nt]);
      v16bf fk = load_b_frag(wk_h + (size_t)nt * 16 * E_, E_, k0);
      acck[nt] = wmma_bf16(a, fk, acck[nt]);
      v16bf fv = load_b_frag(wv_h + (size_t)nt * 16 * E_, E_, k0);
      accv[nt] = wmma_bf16(a, fv, accv[nt]);
    }
  }

#pragma unroll
  for (int nt = 0; nt < 4; ++nt) {
    const int d = nt * 16 + ln;
    const float biasq = bq[h * D_ + d];
    const float biask = bk[h * D_ + d];
    const float biasv = bv[h * D_ + d];
#pragma unroll
    for (int r = 0; r < 8; ++r) {
      const int row = mbase + r + 8 * hi;           // [0, BS)
      const int bb = row / S_, ss = row % S_;
      const size_t qoff = (((size_t)bb * H_ + h) * S_ + ss) * D_ + d;
      q[qoff] = (bf16)((accq[nt][r] + biasq) * 0.125f);  // fold 1/sqrt(64)
      k[qoff] = (bf16)(acck[nt][r] + biask);
      const size_t voff = (((size_t)bb * H_ + h) * D_ + d) * S_ + ss;  // transposed
      vt[voff] = (bf16)(accv[nt][r] + biasv);
    }
  }
}

// ---------------------------------------------------------------- attention
// grid = (B*H, S/64), block = 128 (4 waves, 16 query rows each).
// Flash-style streaming over 64-key tiles with online softmax.
// K tile (64x64) and V^T tile (64x64) are staged once per WG into LDS
// (async global->LDS when available), shared by all 4 waves.
__global__ __launch_bounds__(128) void k_attn(
    const bf16* __restrict__ q, const bf16* __restrict__ k,
    const bf16* __restrict__ vt, bf16* __restrict__ attn) {
  __shared__ bf16 kbuf[64 * 72];     // K tile rows (key, d-contig), stride 72
  __shared__ bf16 vbuf[64 * 72];     // V^T tile rows (d, key-contig), stride 72
  __shared__ bf16 pbuf[4][16 * 72];  // P tile per wave, padded stride 72

  const int bh = blockIdx.x;
  const int wave = threadIdx.x >> 5;
  const int lane = threadIdx.x & 31;
  const int ln = lane & 15, hi = lane >> 4;
  const int b = bh / H_, h = bh % H_;
  const int qbase = blockIdx.y * 64 + wave * 16;

  const bf16* qh = q + (size_t)bh * S_ * D_ + (size_t)qbase * D_;
  const bf16* kh = k + (size_t)bh * S_ * D_;
  const bf16* vth = vt + (size_t)bh * D_ * S_;
  bf16* pl = pbuf[wave];

  // staging coordinates: 128 threads x (16B segment), 16 rows/pass, 4 passes
  const int seg = threadIdx.x & 7;    // 16B segment within a 128B row
  const int srow = threadIdx.x >> 3;  // 0..15

  // Q fragments (reused across all key tiles); score scale folded into Q.
  v16bf aq0 = load_a_frag(qh, D_, 0);
  v16bf aq1 = load_a_frag(qh, D_, 32);

  float mrow[8], lrow[8];
#pragma unroll
  for (int r = 0; r < 8; ++r) { mrow[r] = -3.0e38f; lrow[r] = 0.f; }
  v8f o[4] = {};

  for (int t0 = 0; t0 < S_; t0 += 64) {
    __syncthreads();  // previous tile fully consumed
#pragma unroll
    for (int rr = 0; rr < 4; ++rr) {
      const int row = rr * 16 + srow;  // 0..63
      const bf16* gk = kh + (size_t)(t0 + row) * D_ + seg * 8;
      const bf16* gv = vth + (size_t)row * S_ + t0 + seg * 8;
      bf16* lk = kbuf + row * 72 + seg * 8;
      bf16* lv = vbuf + row * 72 + seg * 8;
#if ASYNC_LDS
      async_copy16(gk, lk);
      async_copy16(gv, lv);
#else
      *(v8bf*)lk = *(const v8bf*)gk;
      *(v8bf*)lv = *(const v8bf*)gv;
#endif
    }
    wait_async();
    __syncthreads();  // tiles visible to all waves

    // ---- scores S = Q K^T (16 x 64) from LDS K tile
    v8f sc[4] = {};
#pragma unroll
    for (int nt = 0; nt < 4; ++nt) {
      const bf16* kb = kbuf + nt * 16 * 72;
      v16bf b0 = load_b_frag(kb, 72, 0);
      sc[nt] = wmma_bf16(aq0, b0, sc[nt]);
      v16bf b1 = load_b_frag(kb, 72, 32);
      sc[nt] = wmma_bf16(aq1, b1, sc[nt]);
    }

    // ---- online softmax (row = r + 8*hi lives within a 16-lane half)
#pragma unroll
    for (int r = 0; r < 8; ++r) {
      float vmax = fmaxf(fmaxf(sc[0][r], sc[1][r]), fmaxf(sc[2][r], sc[3][r]));
      vmax = fmaxf(vmax, __shfl_xor(vmax, 1, 32));
      vmax = fmaxf(vmax, __shfl_xor(vmax, 2, 32));
      vmax = fmaxf(vmax, __shfl_xor(vmax, 4, 32));
      vmax = fmaxf(vmax, __shfl_xor(vmax, 8, 32));
      const float mnew = fmaxf(mrow[r], vmax);
      const float corr = __expf(mrow[r] - mnew);
      mrow[r] = mnew;
      lrow[r] *= corr;
#pragma unroll
      for (int dt = 0; dt < 4; ++dt) o[dt][r] *= corr;

      float rsum = 0.f;
#pragma unroll
      for (int nt = 0; nt < 4; ++nt) {
        const float p = __expf(sc[nt][r] - mnew);
        sc[nt][r] = p;
        rsum += p;
      }
      rsum += __shfl_xor(rsum, 1, 32);
      rsum += __shfl_xor(rsum, 2, 32);
      rsum += __shfl_xor(rsum, 4, 32);
      rsum += __shfl_xor(rsum, 8, 32);
      lrow[r] += rsum;

      // C-layout -> LDS row-major P tile (bf16)
      const int row = r + 8 * hi;
#pragma unroll
      for (int nt = 0; nt < 4; ++nt)
        pl[row * 72 + nt * 16 + ln] = (bf16)sc[nt][r];
    }
    asm volatile("s_wait_dscnt 0x0" ::: "memory");  // our P stores complete

    // ---- O += P * V   (contraction over 64 keys, via LDS V^T tile)
#pragma unroll
    for (int c = 0; c < 2; ++c) {
      v16bf ap = load_a_frag(pl, 72, c * 32);
#pragma unroll
      for (int dt = 0; dt < 4; ++dt) {
        v16bf bv = load_b_frag(vbuf + dt * 16 * 72, 72, c * 32);
        o[dt] = wmma_bf16(ap, bv, o[dt]);
      }
    }
  }

  // ---- epilogue: normalize, store concat-head layout [B*S, E] bf16
#pragma unroll
  for (int dt = 0; dt < 4; ++dt) {
    const int col = h * D_ + dt * 16 + ln;
#pragma unroll
    for (int r = 0; r < 8; ++r) {
      const int sr = qbase + r + 8 * hi;
      attn[((size_t)b * S_ + sr) * E_ + col] = (bf16)(o[dt][r] / lrow[r]);
    }
  }
}

// ---------------------------------------------------------------- out proj
// grid = (BS/128, E/64), block = 256 (8 waves, 16 M-rows each).
__global__ __launch_bounds__(256) void k_proj(
    const bf16* __restrict__ attn, const bf16* __restrict__ wot,
    const float* __restrict__ bo, float* __restrict__ out) {
  const int wave = threadIdx.x >> 5;
  const int lane = threadIdx.x & 31;
  const int ln = lane & 15, hi = lane >> 4;
  const int mbase = blockIdx.x * 128 + wave * 16;
  const int nbase = blockIdx.y * 64;
  const bf16* arow = attn + (size_t)mbase * E_;

  v8f acc[4] = {};
  for (int k0 = 0; k0 < E_; k0 += 32) {
    v16bf a = load_a_frag(arow, E_, k0);
#pragma unroll
    for (int nt = 0; nt < 4; ++nt) {
      v16bf bfrag = load_b_frag(wot + (size_t)(nbase + nt * 16) * E_, E_, k0);
      acc[nt] = wmma_bf16(a, bfrag, acc[nt]);
    }
  }
#pragma unroll
  for (int nt = 0; nt < 4; ++nt) {
    const int col = nbase + nt * 16 + ln;
    const float bias = bo[col];
#pragma unroll
    for (int r = 0; r < 8; ++r) {
      const int row = mbase + r + 8 * hi;
      out[(size_t)row * E_ + col] = acc[nt][r] + bias;
    }
  }
}

// ---------------------------------------------------------------- launcher
extern "C" void kernel_launch(void* const* d_in, const int* in_sizes, int n_in,
                              void* d_out, int out_size, void* d_ws, size_t ws_size,
                              hipStream_t stream) {
  const float* x  = (const float*)d_in[0];
  const float* Wq = (const float*)d_in[1];
  const float* Wk = (const float*)d_in[2];
  const float* Wv = (const float*)d_in[3];
  const float* bq = (const float*)d_in[4];
  const float* bk = (const float*)d_in[5];
  const float* bv = (const float*)d_in[6];
  const float* Wo = (const float*)d_in[7];
  const float* bo = (const float*)d_in[8];
  float* out = (float*)d_out;

  char* ws = (char*)d_ws;
  size_t off = 0;
  auto carve = [&](size_t bytes) -> char* {
    char* p = ws + off;
    off += (bytes + 255) & ~(size_t)255;
    return p;
  };
  bf16* xb   = (bf16*)carve((size_t)BS_ * E_ * sizeof(bf16));      // 12.6 MB
  bf16* wqt  = (bf16*)carve((size_t)H_ * D_ * E_ * sizeof(bf16));  // 1.2 MB
  bf16* wkt  = (bf16*)carve((size_t)H_ * D_ * E_ * sizeof(bf16));
  bf16* wvt  = (bf16*)carve((size_t)H_ * D_ * E_ * sizeof(bf16));
  bf16* wot  = (bf16*)carve((size_t)E_ * E_ * sizeof(bf16));
  bf16* qb   = (bf16*)carve((size_t)BS_ * E_ * sizeof(bf16));
  bf16* kb   = (bf16*)carve((size_t)BS_ * E_ * sizeof(bf16));
  bf16* vtb  = (bf16*)carve((size_t)BS_ * E_ * sizeof(bf16));
  bf16* attn = (bf16*)carve((size_t)BS_ * E_ * sizeof(bf16));
  (void)ws_size; (void)in_sizes; (void)n_in; (void)out_size;

  // Stage 1: convert / transpose
  k_cvt_x<<<1024, 256, 0, stream>>>(x, xb, BS_ * E_);
  k_tr_whead<<<512, 256, 0, stream>>>(Wq, wqt);
  k_tr_whead<<<512, 256, 0, stream>>>(Wk, wkt);
  k_tr_whead<<<512, 256, 0, stream>>>(Wv, wvt);
  k_tr_wo<<<512, 256, 0, stream>>>(Wo, wot);

  // Stage 2: fused QKV projection
  k_qkv<<<dim3(H_, BS_ / 128), 256, 0, stream>>>(xb, wqt, wkt, wvt, bq, bk, bv,
                                                 qb, kb, vtb);

  // Stage 3: flash attention
  k_attn<<<dim3(B_ * H_, S_ / 64), 128, 0, stream>>>(qb, kb, vtb, attn);

  // Stage 4: output projection
  k_proj<<<dim3(BS_ / 128, E_ / 64), 256, 0, stream>>>(attn, wot, bo, out);
}
